// LSTMClassifier_67413806678500
// MI455X (gfx1250) — matrix operand order repaired and checked by reading, not compiled, not run on hardware
//
#include <hip/hip_runtime.h>
#include <hip/hip_bf16.h>

typedef _Float16 v16h __attribute__((ext_vector_type(16)));
typedef _Float16 v4h  __attribute__((ext_vector_type(4)));
typedef float    v8f  __attribute__((ext_vector_type(8)));
typedef float    v4f  __attribute__((ext_vector_type(4)));

#define B_ 512
#define T_ 2048
#define H_ 64
#define G_ 256
#define XCHUNK 32   // timesteps of x staged per LDS refill (layer 0)

__device__ __forceinline__ float fast_sigmoid(float x) {
    float e = __builtin_amdgcn_exp2f(-1.442695041f * x);
    return __builtin_amdgcn_rcpf(1.0f + e);
}
__device__ __forceinline__ float fast_tanh(float x) {
    return 2.0f * fast_sigmoid(2.0f * x) - 1.0f;   // tanh(x) = 2*sigmoid(2x)-1
}

// Light workgroup barrier: only LDS data crosses waves inside the time loop,
// so wait DScnt only (NOT storecnt) -- lets global stores drain in background.
__device__ __forceinline__ void wg_barrier_lds() {
    asm volatile("s_wait_dscnt 0x0\n\t"
                 "s_barrier_signal -1\n\t"
                 "s_barrier_wait -1" ::: "memory");
}

// Async global->LDS copy (8 bytes per lane), tracked by ASYNCcnt.
__device__ __forceinline__ void async_load_b64_to_lds(unsigned lds_off, const void* src) {
    asm volatile("global_load_async_to_lds_b64 %0, %1, off"
                 :: "v"(lds_off), "v"(src) : "memory");
}
__device__ __forceinline__ void wait_async0() {
    asm volatile("s_wait_asynccnt 0x0" ::: "memory");
}

// 16x32 f16 A-fragment (WMMA layout) from row-major [16][64] f16 buffer.
// lanes 0-15: row=lane, K=k0+{0..7,16..23}; lanes 16-31: row=lane-16, K=k0+{8..15,24..31}.
__device__ __forceinline__ v16h load_a_frag(const _Float16* __restrict__ base,
                                            int rowm, int hi, int k0) {
    const _Float16* p = base + rowm * H_ + k0 + hi * 8;
    v16h a;
#pragma unroll
    for (int i = 0; i < 8; ++i) { a[i] = p[i]; a[8 + i] = p[16 + i]; }
    return a;
}

// 32x16 f16 B-fragment (WMMA layout) from row-major fp32 W[G][H]: B[k][n]=W[n][k].
__device__ __forceinline__ v16h load_b_frag(const float* __restrict__ W,
                                            int n, int hi, int k0) {
    const float* src = W + n * H_ + k0 + hi * 8;
    v16h b;
#pragma unroll
    for (int i = 0; i < 8; ++i) { b[i] = (_Float16)src[i]; b[8 + i] = (_Float16)src[16 + i]; }
    return b;
}

#define WMMA_F16(A, Bm, C) \
    __builtin_amdgcn_wmma_f32_16x16x32_f16(false, (A), false, (Bm), (short)0, (C), false, false)

// ---------------------------------------------------------------------------
// Layer 0: input dim 1 (scalar broadcast), recurrent GEMM via WMMA.
// One block = one 16-row batch tile, persistent over all T steps.
// ---------------------------------------------------------------------------
__global__ __launch_bounds__(256, 2) void lstm_layer0(
    const float* __restrict__ x,     // [B, T]
    const float* __restrict__ Wih0,  // [256, 1]
    const float* __restrict__ Whh0,  // [256, 64]
    const float* __restrict__ bih0, const float* __restrict__ bhh0,
    _Float16* __restrict__ out0)     // [T, B, 64] f16
{
    __shared__ _Float16 hA[16 * H_];        // h tile, row-major f16
    __shared__ float    gls[16 * G_];       // gate pre-activations f32
    __shared__ float    xstage[16 * XCHUNK];

    const int tid  = threadIdx.x;
    const int lane = tid & 31;
    const int wave = tid >> 5;               // owns 32 gate columns
    const int r0   = blockIdx.x * 16;
    const int rowm = lane & 15;
    const int hi   = lane >> 4;

    // Resident recurrent-weight B fragments: [tile][kchunk]
    v16h bU[2][2];
#pragma unroll
    for (int t2 = 0; t2 < 2; ++t2)
#pragma unroll
        for (int kc = 0; kc < 2; ++kc)
            bU[t2][kc] = load_b_frag(Whh0, wave * 32 + t2 * 16 + rowm, hi, kc * 32);

    // Elementwise ownership: each thread owns 4 contiguous hidden units of 1 row
    const int erow = tid >> 4;               // 0..15
    const int j0   = (tid & 15) << 2;        // 0,4,...,60
    float wi[16], bs[16];
#pragma unroll
    for (int g = 0; g < 4; ++g)
#pragma unroll
        for (int i = 0; i < 4; ++i) {
            wi[g * 4 + i] = Wih0[g * 64 + j0 + i];
            bs[g * 4 + i] = bih0[g * 64 + j0 + i] + bhh0[g * 64 + j0 + i];
        }
    float cc[4] = {0.f, 0.f, 0.f, 0.f};

    for (int i = tid; i < 16 * H_; i += 256) hA[i] = (_Float16)0.0f;
    __syncthreads();

    const int col0 = wave * 32 + rowm;

    for (int t = 0; t < T_; ++t) {
        // Refill x staging every XCHUNK steps (coalesced burst, amortized)
        if ((t & (XCHUNK - 1)) == 0) {
#pragma unroll
            for (int idx = tid; idx < 16 * XCHUNK; idx += 256) {
                int rw = idx >> 5, tt = idx & (XCHUNK - 1);
                xstage[idx] = x[(size_t)(r0 + rw) * T_ + t + tt];
            }
        }

        v16h a0 = load_a_frag(hA, rowm, hi, 0);
        v16h a1 = load_a_frag(hA, rowm, hi, 32);

        v8f c0 = {}, c1 = {};
        c0 = WMMA_F16(a0, bU[0][0], c0);
        c0 = WMMA_F16(a1, bU[0][1], c0);
        c1 = WMMA_F16(a0, bU[1][0], c1);
        c1 = WMMA_F16(a1, bU[1][1], c1);

#pragma unroll
        for (int v = 0; v < 8; ++v) gls[(v + hi * 8) * G_ + col0]      = c0[v];
#pragma unroll
        for (int v = 0; v < 8; ++v) gls[(v + hi * 8) * G_ + col0 + 16] = c1[v];

        wg_barrier_lds();

        // Elementwise LSTM cell update (gate order i,f,g,o)
        float xv = xstage[erow * XCHUNK + (t & (XCHUNK - 1))];
        v4h  hv;
#pragma unroll
        for (int i = 0; i < 4; ++i) {
            int   jc = j0 + i;
            float gi = gls[erow * G_ + jc]        + xv * wi[0 + i] + bs[0 + i];
            float gf = gls[erow * G_ + 64 + jc]   + xv * wi[4 + i] + bs[4 + i];
            float gg = gls[erow * G_ + 128 + jc]  + xv * wi[8 + i] + bs[8 + i];
            float go = gls[erow * G_ + 192 + jc]  + xv * wi[12 + i] + bs[12 + i];
            gi = fast_sigmoid(gi); gf = fast_sigmoid(gf);
            gg = fast_tanh(gg);    go = fast_sigmoid(go);
            float c = gf * cc[i] + gi * gg;
            cc[i]   = c;
            hv[i]   = (_Float16)(go * fast_tanh(c));
        }
        *(v4h*)&hA[erow * H_ + j0] = hv;                                   // ds b64
        *(v4h*)(out0 + (size_t)t * B_ * H_ + (size_t)(r0 + erow) * H_ + j0) = hv; // global b64

        wg_barrier_lds();
    }
}

// ---------------------------------------------------------------------------
// Layer 1: gates = x_t @ Wih1^T + h @ Whh1^T + bias.  A = [x_t | h], K = 128.
// x_t tiles double-buffered in LDS via async global->LDS copies (ASYNCcnt),
// issued one step ahead and retired just before the publishing barrier.
// ---------------------------------------------------------------------------
__global__ __launch_bounds__(256, 2) void lstm_layer1(
    const _Float16* __restrict__ out0,  // [T, B, 64] f16
    const float* __restrict__ Wih1,     // [256, 64]
    const float* __restrict__ Whh1,     // [256, 64]
    const float* __restrict__ bih1, const float* __restrict__ bhh1,
    float* __restrict__ hlast)          // [B, 64]
{
    __shared__ _Float16 hA[16 * H_];
    __shared__ _Float16 xbuf[2][16 * H_];   // double-buffered x_t tile (2 KB each)
    __shared__ float    gls[16 * G_];

    const int tid  = threadIdx.x;
    const int lane = tid & 31;
    const int wave = tid >> 5;
    const int r0   = blockIdx.x * 16;
    const int rowm = lane & 15;
    const int hi   = lane >> 4;

    v16h bW[2][2], bU[2][2];
#pragma unroll
    for (int t2 = 0; t2 < 2; ++t2)
#pragma unroll
        for (int kc = 0; kc < 2; ++kc) {
            int n      = wave * 32 + t2 * 16 + rowm;
            bW[t2][kc] = load_b_frag(Wih1, n, hi, kc * 32);
            bU[t2][kc] = load_b_frag(Whh1, n, hi, kc * 32);
        }

    const int erow = tid >> 4;
    const int j0   = (tid & 15) << 2;
    float bs[16];
#pragma unroll
    for (int g = 0; g < 4; ++g)
#pragma unroll
        for (int i = 0; i < 4; ++i)
            bs[g * 4 + i] = bih1[g * 64 + j0 + i] + bhh1[g * 64 + j0 + i];
    float cc[4] = {0.f, 0.f, 0.f, 0.f};

    for (int i = tid; i < 16 * H_; i += 256) hA[i] = (_Float16)0.0f;

    // Each thread copies 8 contiguous bytes of the 2 KB x_t tile.
    const unsigned xb_lds = (unsigned)(uintptr_t)&xbuf[0][0] + (unsigned)tid * 8u;
    const char*    gbase  = (const char*)out0 + (size_t)r0 * H_ * 2 + (size_t)tid * 8;

    async_load_b64_to_lds(xb_lds, gbase);        // t = 0 tile
    wait_async0();
    __syncthreads();

    const int col0 = wave * 32 + rowm;

    for (int t = 0; t < T_; ++t) {
        const _Float16* xb = xbuf[t & 1];
        v16h ax0 = load_a_frag(xb, rowm, hi, 0);
        v16h ax1 = load_a_frag(xb, rowm, hi, 32);
        v16h ah0 = load_a_frag(hA, rowm, hi, 0);
        v16h ah1 = load_a_frag(hA, rowm, hi, 32);

        if (t + 1 < T_)   // prefetch next x_t tile into the other buffer
            async_load_b64_to_lds(xb_lds + (unsigned)((t + 1) & 1) * 2048u,
                                  gbase + (size_t)(t + 1) * B_ * H_ * 2);

        v8f c0 = {}, c1 = {};
        c0 = WMMA_F16(ax0, bW[0][0], c0);
        c0 = WMMA_F16(ax1, bW[0][1], c0);
        c0 = WMMA_F16(ah0, bU[0][0], c0);
        c0 = WMMA_F16(ah1, bU[0][1], c0);
        c1 = WMMA_F16(ax0, bW[1][0], c1);
        c1 = WMMA_F16(ax1, bW[1][1], c1);
        c1 = WMMA_F16(ah0, bU[1][0], c1);
        c1 = WMMA_F16(ah1, bU[1][1], c1);

#pragma unroll
        for (int v = 0; v < 8; ++v) gls[(v + hi * 8) * G_ + col0]      = c0[v];
#pragma unroll
        for (int v = 0; v < 8; ++v) gls[(v + hi * 8) * G_ + col0 + 16] = c1[v];

        wg_barrier_lds();

        v4h hv; v4f hf;
#pragma unroll
        for (int i = 0; i < 4; ++i) {
            int   jc = j0 + i;
            float gi = gls[erow * G_ + jc]       + bs[0 + i];
            float gf = gls[erow * G_ + 64 + jc]  + bs[4 + i];
            float gg = gls[erow * G_ + 128 + jc] + bs[8 + i];
            float go = gls[erow * G_ + 192 + jc] + bs[12 + i];
            gi = fast_sigmoid(gi); gf = fast_sigmoid(gf);
            gg = fast_tanh(gg);    go = fast_sigmoid(go);
            float c = gf * cc[i] + gi * gg;
            cc[i]   = c;
            float h = go * fast_tanh(c);
            hv[i] = (_Float16)h;
            hf[i] = h;
        }
        *(v4h*)&hA[erow * H_ + j0] = hv;
        if (t == T_ - 1)
            *(v4f*)&hlast[(size_t)(r0 + erow) * H_ + j0] = hf;

        wait_async0();       // retire t+1 tile before publishing barrier
        wg_barrier_lds();
    }
}

// ---------------------------------------------------------------------------
// Head: relu(h @ W1^T + b1) @ W2^T + b2  -> [B]
// ---------------------------------------------------------------------------
__global__ __launch_bounds__(256) void head_kernel(
    const float* __restrict__ hlast, const float* __restrict__ W1,
    const float* __restrict__ b1, const float* __restrict__ W2,
    const float* __restrict__ b2, float* __restrict__ out)
{
    __shared__ float sW1[32 * 64];
    __shared__ float sb1[32], sW2[32];
    const int tid = threadIdx.x;
    for (int i = tid; i < 32 * 64; i += 256) sW1[i] = W1[i];
    if (tid < 32) { sb1[tid] = b1[tid]; sW2[tid] = W2[tid]; }
    __syncthreads();

    const int row = blockIdx.x * 256 + tid;
    float hv[64];
#pragma unroll
    for (int i = 0; i < 64; ++i) hv[i] = hlast[(size_t)row * 64 + i];

    float acc = b2[0];
#pragma unroll 4
    for (int m = 0; m < 32; ++m) {
        float s = sb1[m];
#pragma unroll
        for (int i = 0; i < 64; ++i) s = fmaf(hv[i], sW1[m * 64 + i], s);
        s = fmaxf(s, 0.0f);
        acc = fmaf(s, sW2[m], acc);
    }
    out[row] = acc;
}

extern "C" void kernel_launch(void* const* d_in, const int* in_sizes, int n_in,
                              void* d_out, int out_size, void* d_ws, size_t ws_size,
                              hipStream_t stream) {
    const float* x    = (const float*)d_in[0];
    const float* Wih0 = (const float*)d_in[1];
    const float* Whh0 = (const float*)d_in[2];
    const float* bih0 = (const float*)d_in[3];
    const float* bhh0 = (const float*)d_in[4];
    const float* Wih1 = (const float*)d_in[5];
    const float* Whh1 = (const float*)d_in[6];
    const float* bih1 = (const float*)d_in[7];
    const float* bhh1 = (const float*)d_in[8];
    const float* W1   = (const float*)d_in[9];
    const float* b1   = (const float*)d_in[10];
    const float* W2   = (const float*)d_in[11];
    const float* b2   = (const float*)d_in[12];

    _Float16* out0  = (_Float16*)d_ws;
    float*    hlast = (float*)((char*)d_ws + (size_t)T_ * B_ * H_ * sizeof(_Float16));

    lstm_layer0<<<B_ / 16, 256, 0, stream>>>(x, Wih0, Whh0, bih0, bhh0, out0);
    lstm_layer1<<<B_ / 16, 256, 0, stream>>>(out0, Wih1, Whh1, bih1, bhh1, hlast);
    head_kernel<<<B_ / 256, 256, 0, stream>>>(hlast, W1, b1, W2, b2, (float*)d_out);
}
